// SineLSTM_52578989638068
// MI455X (gfx1250) — matrix operand order repaired and checked by reading, not compiled, not run on hardware
//
#include <hip/hip_runtime.h>

// ---------------- problem constants (fixed by reference) ----------------
#define HH   512          // hidden size
#define BB   128          // batch
#define ROWS 16           // batch rows per workgroup (one WMMA M-tile)
#define NWG  (BB / ROWS)  // 8 persistent workgroups, fully independent
#define NTHR 256          // 8 wave32 waves per workgroup

typedef __attribute__((ext_vector_type(16))) __bf16 v16bf;
typedef __attribute__((ext_vector_type(8)))  float  v8f;

struct Frag { union { uint4 u[2]; v16bf v; }; };

// ---------------- LDS layout (dynamic shared, > 64KB) ----------------
constexpr int LDH       = 520;               // bf16 row stride (16B-aligned rows, bank-staggered)
constexpr int LDF       = 516;               // f32 row stride (stagger 8-row bank aliasing)
constexpr int HB        = ROWS * LDH;        // bf16 elems per h buffer
constexpr int H1B_OFF   = 0;
constexpr int HB_BYTES  = 2 * HB * 2;        // double-buffered, 2B/elem
constexpr int H2B_OFF   = H1B_OFF + HB_BYTES;
constexpr int H2F_OFF   = H2B_OFF + HB_BYTES;
constexpr int F32_BYTES = ROWS * LDF * 4;
constexpr int C1_OFF    = H2F_OFF + F32_BYTES;   // layer-1 cell state (f32, LDS-resident)
constexpr int C2_OFF    = C1_OFF + F32_BYTES;    // layer-2 cell state
constexpr int B1_OFF    = C2_OFF + F32_BYTES;    // b_ih1+b_hh1 summed, 4H f32
constexpr int B2_OFF    = B1_OFF + 4 * HH * 4;   // b_ih2+b_hh2 summed
constexpr int WI1_OFF   = B2_OFF + 4 * HH * 4;   // W_ih1 column (rank-1 input weight)
constexpr int WO_OFF    = WI1_OFF + 4 * HH * 4;  // W_out row
constexpr int XV_OFF    = WO_OFF + HH * 4;       // per-row step input
constexpr int SMEM_BYTES = XV_OFF + ROWS * 4;    // ~192 KB, fits 320 KB WGP LDS

constexpr int WELEMS = 4 * HH * HH;          // elements per recurrent weight matrix

// Branch-free activations using the raw hardware exp2 (no libm denorm guard: LSTM gate
// pre-activations are O(10), nowhere near the v_exp_f32 flush range at -126).
// sigmf lowers to exactly: v_mul + v_exp_f32 + v_add + v_rcp_f32; tanh_fast adds a v_fma.
__device__ __forceinline__ float sigmf(float x) {
  return __builtin_amdgcn_rcpf(1.0f + __builtin_amdgcn_exp2f(x * -1.442695040888963f));
}
__device__ __forceinline__ float tanh_fast(float x) {
  // tanh(x) = 2*sigmoid(2x) - 1
  return fmaf(2.0f,
              __builtin_amdgcn_rcpf(1.0f + __builtin_amdgcn_exp2f(x * -2.885390081777927f)),
              -1.0f);
}

// Accumulate acc[q] += h(16 x K=512) @ W[q*512 + ncol-tile]^T  (one 16-wide column tile
// per gate quarter q). A/B fragment layouts per CDNA5 ISA 16-bit WMMA tables.
__device__ __forceinline__ void gemm_k512(v8f acc[4],
                                          const __bf16* __restrict__ hbuf,  // LDS, stride LDH
                                          const __bf16* __restrict__ W,     // global bf16 [2048][512]
                                          int ncol, int lr, int gh) {
#pragma clang loop unroll_count(2)
  for (int k = 0; k < HH; k += 32) {
    Frag a;
    const __bf16* ap = hbuf + lr * LDH + k + gh * 8;
    a.u[0] = *(const uint4*)ap;
    a.u[1] = *(const uint4*)(ap + 16);
#pragma unroll
    for (int q = 0; q < 4; ++q) {
      Frag b;
      const __bf16* bp = W + (size_t)(q * HH + ncol) * HH + k + gh * 8;
      b.u[0] = *(const uint4*)bp;
      b.u[1] = *(const uint4*)(bp + 16);
      acc[q] = __builtin_amdgcn_wmma_f32_16x16x32_bf16(
          false, a.v, false, b.v, (short)0, acc[q], false, false);
    }
  }
}

// ---------------- f32 -> bf16 one-time weight conversion ----------------
__global__ void cvt_weights_bf16(const float* __restrict__ whh1,
                                 const float* __restrict__ wih2,
                                 const float* __restrict__ whh2,
                                 __bf16* __restrict__ o) {
  int stride = gridDim.x * blockDim.x;
  for (int i = blockIdx.x * blockDim.x + threadIdx.x; i < WELEMS; i += stride) {
    o[i]              = (__bf16)whh1[i];
    o[i + WELEMS]     = (__bf16)wih2[i];
    o[i + 2 * WELEMS] = (__bf16)whh2[i];
  }
}

// ---------------- persistent 2-layer LSTM, batch-partitioned ----------------
__global__ __launch_bounds__(NTHR, 1) void lstm_persist(
    const float* __restrict__ x,      // [B,T]
    const float* __restrict__ wih1,   // [4H]
    const float* __restrict__ bih1, const float* __restrict__ bhh1,
    const float* __restrict__ bih2, const float* __restrict__ bhh2,
    const float* __restrict__ wout,   // [H]
    const float* __restrict__ bo,     // [1]
    const __bf16* __restrict__ Whh1b, // [4H][H] bf16
    const __bf16* __restrict__ Wih2b,
    const __bf16* __restrict__ Whh2b,
    float* __restrict__ out,          // [B, S]
    int T, int S) {
  extern __shared__ char smem[];
  __bf16* h1b    = (__bf16*)(smem + H1B_OFF);   // 2 x [16][LDH]
  __bf16* h2b    = (__bf16*)(smem + H2B_OFF);
  float*  h2f    = (float*)(smem + H2F_OFF);    // [16][LDF] f32 h2 for output head
  float*  c1s    = (float*)(smem + C1_OFF);     // [16][LDF] layer-1 cell state
  float*  c2s    = (float*)(smem + C2_OFF);     // [16][LDF] layer-2 cell state
  float*  bias1s = (float*)(smem + B1_OFF);
  float*  bias2s = (float*)(smem + B2_OFF);
  float*  wih1s  = (float*)(smem + WI1_OFF);
  float*  wouts  = (float*)(smem + WO_OFF);
  float*  xv     = (float*)(smem + XV_OFF);     // per-row input this step

  const int tid  = threadIdx.x;
  const int wg   = blockIdx.x;
  const int wid  = tid >> 5;        // wave id 0..7
  const int lane = tid & 31;
  const int gh   = lane >> 4;       // lane half
  const int lr   = lane & 15;

  // init: zero h buffers + cell state, stage biases / small weights
  {
    unsigned int* z = (unsigned int*)smem;
    for (int i = tid; i < (2 * HB_BYTES) / 4; i += NTHR) z[i] = 0u;   // h1b,h2b
    unsigned int* zc = (unsigned int*)(smem + C1_OFF);
    for (int i = tid; i < (2 * F32_BYTES) / 4; i += NTHR) zc[i] = 0u; // c1s,c2s
    for (int i = tid; i < 4 * HH; i += NTHR) {
      bias1s[i] = bih1[i] + bhh1[i];
      bias2s[i] = bih2[i] + bhh2[i];
      wih1s[i]  = wih1[i];
    }
    for (int i = tid; i < HH; i += NTHR) wouts[i] = wout[i];
    if (tid < ROWS) xv[tid] = x[(size_t)(wg * ROWS + tid) * T];
  }
  const float bout = bo[0];
  __syncthreads();

  int cur = 0;
  for (int t = 0; t < S; ++t) {
    const int nxt = cur ^ 1;
    const __bf16* h1cur = h1b + cur * HB;
    const __bf16* h2cur = h2b + cur * HB;
    __bf16* h1nxt = h1b + nxt * HB;
    __bf16* h2nxt = h2b + nxt * HB;

    float xr[8];
#pragma unroll
    for (int r = 0; r < 8; ++r) xr[r] = xv[r + 8 * gh];

    // ---- phase 1: layer-1 gates = x*w_ih1 + b + h1_prev @ W_hh1^T, then cellwise ----
    for (int p = 0; p < 4; ++p) {
      const int npos = wid + p * 8;
      const int ncol = npos * 16 + lr;       // hidden index within each gate quarter
      v8f acc[4];
#pragma unroll
      for (int q = 0; q < 4; ++q) {
        const float b = bias1s[q * HH + ncol];
        const float w = wih1s[q * HH + ncol];
#pragma unroll
        for (int r = 0; r < 8; ++r) acc[q][r] = fmaf(xr[r], w, b);
      }
      gemm_k512(acc, h1cur, Whh1b, ncol, lr, gh);
#pragma unroll
      for (int r = 0; r < 8; ++r) {
        const int ci = (r + 8 * gh) * LDF + ncol;
        const float gi = sigmf(acc[0][r]);
        const float gf = sigmf(acc[1][r]);
        const float gg = tanh_fast(acc[2][r]);
        const float go = sigmf(acc[3][r]);
        const float c  = fmaf(gf, c1s[ci], gi * gg);
        c1s[ci] = c;
        const float h = go * tanh_fast(c);
        h1nxt[(r + 8 * gh) * LDH + ncol] = (__bf16)h;
      }
    }
    __syncthreads();

    // ---- phase 2: layer-2 gates = b + h2_prev @ W_hh2^T + h1_new @ W_ih2^T ----
    for (int p = 0; p < 4; ++p) {
      const int npos = wid + p * 8;
      const int ncol = npos * 16 + lr;
      v8f acc[4];
#pragma unroll
      for (int q = 0; q < 4; ++q) {
        const float b = bias2s[q * HH + ncol];
#pragma unroll
        for (int r = 0; r < 8; ++r) acc[q][r] = b;
      }
      gemm_k512(acc, h2cur, Whh2b, ncol, lr, gh);
      gemm_k512(acc, h1nxt, Wih2b, ncol, lr, gh);
#pragma unroll
      for (int r = 0; r < 8; ++r) {
        const int ci = (r + 8 * gh) * LDF + ncol;
        const float gi = sigmf(acc[0][r]);
        const float gf = sigmf(acc[1][r]);
        const float gg = tanh_fast(acc[2][r]);
        const float go = sigmf(acc[3][r]);
        const float c  = fmaf(gf, c2s[ci], gi * gg);
        c2s[ci] = c;
        const float h = go * tanh_fast(c);
        h2nxt[(r + 8 * gh) * LDH + ncol] = (__bf16)h;
        h2f[ci] = h;
      }
    }
    __syncthreads();

    // ---- phase 3: out = h2 @ W_out^T + b_out ; feed x for future steps ----
#pragma unroll
    for (int rr = 0; rr < 2; ++rr) {
      const int m = wid * 2 + rr;
      float s = 0.0f;
      for (int kk = lane; kk < HH; kk += 32) s = fmaf(h2f[m * LDF + kk], wouts[kk], s);
#pragma unroll
      for (int off = 16; off > 0; off >>= 1) s += __shfl_xor(s, off, 32);
      if (lane == 0) {
        const float ov = s + bout;
        const int rowg = wg * ROWS + m;
        out[(size_t)rowg * S + t] = ov;
        xv[m] = (t + 1 < T) ? x[(size_t)rowg * T + (t + 1)] : ov;
      }
    }
    __syncthreads();
    cur = nxt;
  }
}

extern "C" void kernel_launch(void* const* d_in, const int* in_sizes, int n_in,
                              void* d_out, int out_size, void* d_ws, size_t ws_size,
                              hipStream_t stream) {
  const float* x    = (const float*)d_in[0];
  // d_in[1] = future (derived from out_size instead)
  const float* wih1 = (const float*)d_in[2];
  const float* whh1 = (const float*)d_in[3];
  const float* bih1 = (const float*)d_in[4];
  const float* bhh1 = (const float*)d_in[5];
  const float* wih2 = (const float*)d_in[6];
  const float* whh2 = (const float*)d_in[7];
  const float* bih2 = (const float*)d_in[8];
  const float* bhh2 = (const float*)d_in[9];
  const float* wout = (const float*)d_in[10];
  const float* bo   = (const float*)d_in[11];
  float* out = (float*)d_out;
  __bf16* wsb = (__bf16*)d_ws;                 // 3 x 1M bf16 = 6 MB scratch

  const int T = in_sizes[0] / BB;              // 512
  const int S = out_size / BB;                 // T + future = 576

  hipFuncSetAttribute((const void*)lstm_persist,
                      hipFuncAttributeMaxDynamicSharedMemorySize, SMEM_BYTES);

  cvt_weights_bf16<<<512, 256, 0, stream>>>(whh1, wih2, whh2, wsb);
  lstm_persist<<<NWG, NTHR, SMEM_BYTES, stream>>>(
      x, wih1, bih1, bhh1, bih2, bhh2, wout, bo,
      wsb, wsb + WELEMS, wsb + 2 * WELEMS, out, T, S);
}
